// NCA_model_34471407517754
// MI455X (gfx1250) — matrix operand order repaired and checked by reading, not compiled, not run on hardware
//
#include <hip/hip_runtime.h>

typedef __attribute__((ext_vector_type(16))) _Float16 v16h;
typedef __attribute__((ext_vector_type(8)))  _Float16 v8h;
typedef __attribute__((ext_vector_type(8)))  float    v8f;

#define Bn   32
#define Cn   16
#define HIDN 128
#define Gn   256
#define XSTR 20           // padded row stride of the LDS x tile (18 used)
#define PLANE (18*XSTR)   // 360 floats per channel plane

#define AS1 __attribute__((address_space(1)))
#define AS3 __attribute__((address_space(3)))

#if __has_builtin(__builtin_amdgcn_global_load_async_to_lds_b32) && \
    __has_builtin(__builtin_amdgcn_s_wait_asynccnt)
#define HAVE_ASYNC 1
#else
#define HAVE_ASYNC 0
#warning "gfx1250 async-to-LDS builtins unavailable; using synchronous staging"
#endif

// Depthwise 3x3 perception filter, cross-correlation (XLA conv semantics):
// j==0: identity, j==1: sobel_x/8, j==2: sobel_y/8
__device__ __forceinline__ float conv_y(const float* cen, int k) {
  const int c = k / 3;
  const int j = k - 3 * c;
  const float* p = cen + c * PLANE;
  if (j == 0) return p[0];
  float a00 = p[-XSTR - 1], a02 = p[-XSTR + 1];
  float a20 = p[ XSTR - 1], a22 = p[ XSTR + 1];
  if (j == 1) {
    float a10 = p[-1], a12 = p[1];
    return ((a02 - a00) + 2.f * (a12 - a10) + (a22 - a20)) * 0.125f;
  }
  float a01 = p[-XSTR], a21 = p[XSTR];
  return ((a20 - a00) + 2.f * (a21 - a01) + (a22 - a02)) * 0.125f;
}

__device__ __forceinline__ unsigned int pack2(float a, float b) {
  union { _Float16 h[2]; unsigned int u; } t;
  t.h[0] = (_Float16)a; t.h[1] = (_Float16)b;
  return t.u;
}

__global__ __launch_bounds__(256) void nca_update_kernel(
    const float* __restrict__ x, const float* __restrict__ w1,
    const float* __restrict__ b1, const float* __restrict__ w2,
    const float* __restrict__ um, float* __restrict__ out,
    float* __restrict__ alpha_new)
{
  __shared__ __align__(16) float     xs[Cn * PLANE];    // x tile + halo, f32
  __shared__ __align__(16) _Float16  w1s[HIDN * 64];    // w1 f16, K padded to 64
  __shared__ __align__(16) _Float16  w2s[Cn * HIDN];    // w2 f16
  __shared__ __align__(32) float     b1s[HIDN];

  const int tx = blockIdx.x, ty = blockIdx.y, b = blockIdx.z;
  const int tid = threadIdx.x;
  const int gx0 = tx * 16, gy0 = ty * 16;

  // ---- stage x tile (18x18 halo, 16 channels), zero-fill out of bounds ----
  // One (ly,lx) position per thread iteration; unrolled channel loop.
  for (int pos = tid; pos < 18 * 18; pos += 256) {
    const int ly = pos / 18;
    const int lx = pos - ly * 18;
    const int gx = gx0 - 1 + lx, gy = gy0 - 1 + ly;
    const int loff = ly * XSTR + lx;
    const bool inb = (gx >= 0 && gx < Gn && gy >= 0 && gy < Gn);
#if HAVE_ASYNC
    if (inb) {
      const float* gp = x + ((size_t)(b * Cn) * Gn + gy) * Gn + gx;
      #pragma unroll
      for (int c = 0; c < Cn; ++c)
        __builtin_amdgcn_global_load_async_to_lds_b32(
            (AS1 int*)(gp + (size_t)c * Gn * Gn),
            (AS3 int*)(xs + c * PLANE + loff), 0, 0);
    } else {
      #pragma unroll
      for (int c = 0; c < Cn; ++c) xs[c * PLANE + loff] = 0.f;  // disjoint addrs
    }
#else
    const float* gp = x + ((size_t)(b * Cn) * Gn + gy) * Gn + gx;
    #pragma unroll
    for (int c = 0; c < Cn; ++c)
      xs[c * PLANE + loff] = inb ? gp[(size_t)c * Gn * Gn] : 0.f;
#endif
  }
  // ---- stage weights as f16 (needs conversion -> synchronous) ----
  for (int idx = tid; idx < HIDN * 64; idx += 256) {
    int m = idx >> 6, k = idx & 63;
    w1s[idx] = (k < 48) ? (_Float16)w1[m * 48 + k] : (_Float16)0.f;
  }
  for (int idx = tid; idx < Cn * HIDN; idx += 256)
    w2s[idx] = (_Float16)w2[idx];
  for (int idx = tid; idx < HIDN; idx += 256)
    b1s[idx] = b1[idx];
#if HAVE_ASYNC
  __builtin_amdgcn_s_wait_asynccnt(0);
#endif
  __syncthreads();

  const int wave = tid >> 5;
  const int lane = tid & 31;
  const int pix  = lane & 15;
  const bool hih = (lane & 16) != 0;   // high half-wave
  const int koff = hih ? 8 : 0;
  const int xaddr = (lane ^ 16) << 2;  // ds_bpermute half-wave swap

  for (int g = 0; g < 2; ++g) {
    const int py = wave * 2 + g;              // tile row (0..15)
    const int lx = pix + 1, ly = py + 1;      // LDS coords
    const float* cen = xs + ly * XSTR + lx;   // channel-0 center

    // ---- build B operands (perception vector in WMMA B layout) ----
    // lanes 0-15: element e = K=e of pixel `lane`; lanes 16-31: K=16+e
    v16h bOp0, bOp1;
    if (!hih) {
      #pragma unroll
      for (int i = 0; i < 16; ++i) bOp0[i] = (_Float16)conv_y(cen, i);
      #pragma unroll
      for (int i = 0; i < 16; ++i) bOp1[i] = (_Float16)conv_y(cen, 32 + i);
    } else {
      #pragma unroll
      for (int i = 0; i < 16; ++i) bOp0[i] = (_Float16)conv_y(cen, 16 + i);
      #pragma unroll
      for (int i = 0; i < 16; ++i) bOp1[i] = (_Float16)0.f;   // K=48..63 pad
    }

    // ---- layer 1: h(128 x 16px) = W1(128x48pad64) x Y + b1 ----
    // Bias folded into the WMMA C operand (D layout: lane<16 -> M=r, else M=8+r)
    v8f acc[8];
    #pragma unroll
    for (int t = 0; t < 8; ++t)
      acc[t] = *(const v8f*)(b1s + t * 16 + koff);

    #pragma unroll
    for (int t = 0; t < 8; ++t) {
      const _Float16* row = w1s + (t * 16 + pix) * 64;
      v8h lo0 = *(const v8h*)(row + 2 * koff);
      v8h hi0 = *(const v8h*)(row + 8 + 2 * koff);
      v8h lo1 = *(const v8h*)(row + 32 + 2 * koff);
      v8h hi1 = *(const v8h*)(row + 40 + 2 * koff);
      v16h A0, A1;
      #pragma unroll
      for (int i = 0; i < 8; ++i) {
        A0[i] = lo0[i]; A0[8 + i] = hi0[i];
        A1[i] = lo1[i]; A1[8 + i] = hi1[i];
      }
      acc[t] = __builtin_amdgcn_wmma_f32_16x16x32_f16(
          false, A0, false, bOp0, (short)0, acc[t], false, false);
      acc[t] = __builtin_amdgcn_wmma_f32_16x16x32_f16(
          false, A1, false, bOp1, (short)0, acc[t], false, false);
    }

    // ---- relu + pack to f16 pairs ----
    unsigned int hpk[8][4];
    #pragma unroll
    for (int t = 0; t < 8; ++t) {
      #pragma unroll
      for (int r = 0; r < 8; ++r) {
        float v = acc[t][r];
        acc[t][r] = v > 0.f ? v : 0.f;
      }
      #pragma unroll
      for (int r = 0; r < 4; ++r)
        hpk[t][r] = pack2(acc[t][2 * r], acc[t][2 * r + 1]);
    }

    // ---- layer 2: dx(16 x 16px) = W2(16x128) x h ----
    // Re-layout h (D-layout) into B-layout via half-wave swap (ds_bpermute lane^16)
    v8f acc2;
    #pragma unroll
    for (int r = 0; r < 8; ++r) acc2[r] = 0.f;

    #pragma unroll
    for (int t2 = 0; t2 < 4; ++t2) {
      unsigned int blo[4], bhi[4];
      #pragma unroll
      for (int r = 0; r < 4; ++r) {
        unsigned int swA = (unsigned int)__builtin_amdgcn_ds_bpermute(xaddr, (int)hpk[2 * t2][r]);
        unsigned int swB = (unsigned int)__builtin_amdgcn_ds_bpermute(xaddr, (int)hpk[2 * t2 + 1][r]);
        blo[r] = hih ? swB : hpk[2 * t2][r];       // K = 32*t2 + (0..7 | 16..23)
        bhi[r] = hih ? hpk[2 * t2 + 1][r] : swA;   // K = 32*t2 + (8..15 | 24..31)
      }
      v16h B2;
      #pragma unroll
      for (int r = 0; r < 4; ++r) {
        union { unsigned int u; _Float16 h[2]; } a, c;
        a.u = blo[r]; c.u = bhi[r];
        B2[2 * r]     = a.h[0]; B2[2 * r + 1]     = a.h[1];
        B2[8 + 2 * r] = c.h[0]; B2[8 + 2 * r + 1] = c.h[1];
      }
      const _Float16* row2 = w2s + pix * HIDN + t2 * 32;
      v8h lo = *(const v8h*)(row2 + 2 * koff);
      v8h hh = *(const v8h*)(row2 + 8 + 2 * koff);
      v16h A2;
      #pragma unroll
      for (int i = 0; i < 8; ++i) { A2[i] = lo[i]; A2[8 + i] = hh[i]; }
      acc2 = __builtin_amdgcn_wmma_f32_16x16x32_f16(
          false, A2, false, B2, (short)0, acc2, false, false);
    }

    // ---- x_new = x + dx*mask; write out + new alpha plane ----
    const int gx = gx0 + pix, gy = gy0 + py;
    const float mk = um[((size_t)b * Gn + gy) * Gn + gx];
    #pragma unroll
    for (int r = 0; r < 8; ++r) {
      int c = koff + r;                       // output channel (D layout)
      float xv = xs[c * PLANE + ly * XSTR + lx];
      float xn = xv + acc2[r] * mk;
      out[((size_t)(b * Cn + c) * Gn + gy) * Gn + gx] = xn;
      if (c == 3)
        alpha_new[((size_t)b * Gn + gy) * Gn + gx] = xn;
    }
  }
}

__global__ __launch_bounds__(256) void nca_life_kernel(
    const float* __restrict__ x, const float* __restrict__ alpha_new,
    float* __restrict__ out)
{
  __shared__ __align__(16) float ao[PLANE];
  __shared__ __align__(16) float an[PLANE];
  const int tx = blockIdx.x, ty = blockIdx.y, b = blockIdx.z;
  const int tid = threadIdx.x;
  const int gx0 = tx * 16, gy0 = ty * 16;

  for (int pos = tid; pos < 18 * 18; pos += 256) {
    const int ly = pos / 18;
    const int lx = pos - ly * 18;
    const int gx = gx0 - 1 + lx, gy = gy0 - 1 + ly;
    const int loff = ly * XSTR + lx;
    const bool inb = (gx >= 0 && gx < Gn && gy >= 0 && gy < Gn);
#if HAVE_ASYNC
    if (inb) {
      __builtin_amdgcn_global_load_async_to_lds_b32(
          (AS1 int*)(x + ((size_t)(b * Cn + 3) * Gn + gy) * Gn + gx),
          (AS3 int*)(ao + loff), 0, 0);
      __builtin_amdgcn_global_load_async_to_lds_b32(
          (AS1 int*)(alpha_new + ((size_t)b * Gn + gy) * Gn + gx),
          (AS3 int*)(an + loff), 0, 0);
    } else {
      ao[loff] = 0.f;
      an[loff] = 0.f;
    }
#else
    float vo = 0.f, vn = 0.f;
    if (inb) {
      vo = x[((size_t)(b * Cn + 3) * Gn + gy) * Gn + gx];
      vn = alpha_new[((size_t)b * Gn + gy) * Gn + gx];
    }
    ao[loff] = vo;
    an[loff] = vn;
#endif
  }
#if HAVE_ASYNC
  __builtin_amdgcn_s_wait_asynccnt(0);
#endif
  __syncthreads();

  const int px = tid & 15, py = tid >> 4;
  const int lx = px + 1, ly = py + 1;
  float mo = -1e30f, mn = -1e30f;
  #pragma unroll
  for (int dy = -1; dy <= 1; ++dy) {
    #pragma unroll
    for (int dx = -1; dx <= 1; ++dx) {
      mo = fmaxf(mo, ao[(ly + dy) * XSTR + lx + dx]);
      mn = fmaxf(mn, an[(ly + dy) * XSTR + lx + dx]);
    }
  }
  const float life = (mo > 0.1f && mn > 0.1f) ? 1.f : 0.f;
  const int gx = gx0 + px, gy = gy0 + py;
  #pragma unroll
  for (int c = 0; c < Cn; ++c) {
    size_t idx = ((size_t)(b * Cn + c) * Gn + gy) * Gn + gx;
    out[idx] = out[idx] * life;
  }
}

extern "C" void kernel_launch(void* const* d_in, const int* in_sizes, int n_in,
                              void* d_out, int out_size, void* d_ws, size_t ws_size,
                              hipStream_t stream) {
  (void)in_sizes; (void)n_in; (void)out_size; (void)ws_size;
  const float* x  = (const float*)d_in[0];
  const float* w1 = (const float*)d_in[1];
  const float* b1 = (const float*)d_in[2];
  const float* w2 = (const float*)d_in[3];
  const float* um = (const float*)d_in[4];
  float* out = (float*)d_out;
  float* alpha_new = (float*)d_ws;   // B*G*G floats = 8 MB of scratch

  dim3 grid(Gn / 16, Gn / 16, Bn);
  dim3 blk(256);
  nca_update_kernel<<<grid, blk, 0, stream>>>(x, w1, b1, w2, um, out, alpha_new);
  nca_life_kernel<<<grid, blk, 0, stream>>>(x, alpha_new, out);
}
